// GCNet_11433202942399
// MI455X (gfx1250) — matrix operand order repaired and checked by reading, not compiled, not run on hardware
//
#include <hip/hip_runtime.h>
#include <hip/hip_bf16.h>
#include <math.h>

typedef __attribute__((ext_vector_type(16))) _Float16 v16h;
typedef __attribute__((ext_vector_type(8)))  _Float16 v8h;
typedef __attribute__((ext_vector_type(8)))  float    v8f;

#define TS 64      // intermediate tile row stride (halves): 16 rows x 64 cols
#define WAVES 4    // waves per block (128 threads)

// Intra-wave LDS fence: same-wave DS ops retire in order; DScnt==0 means all
// prior ds_store/ds_load done. The "memory" clobber stops compiler reordering.
#define WSYNC() asm volatile("s_wait_dscnt 0" ::: "memory")

__device__ __forceinline__ float fast_tanh(float v) {
#if __has_builtin(__builtin_amdgcn_tanhf)
  return __builtin_amdgcn_tanhf(v);   // gfx1250 v_tanh_f32
#else
  return tanhf(v);
#endif
}

// A-matrix 16x32 f16 fragment (ISA 7.12.2):
// lanes 0-15:  M=lane,    halves[0..7]=K0..7,  halves[8..15]=K16..23
// lanes 16-31: M=lane-16, halves[0..7]=K8..15, halves[8..15]=K24..31
__device__ __forceinline__ v16h load_A_frag(const _Float16* tile, int lane, int kChunk) {
  const int m = lane & 15;
  const _Float16* p = tile + m * TS + kChunk * 32 + ((lane & 16) ? 8 : 0);
  union { v16h v; v8h h[2]; } u;
  u.h[0] = *(const v8h*)(p);        // ds_load_b128
  u.h[1] = *(const v8h*)(p + 16);   // ds_load_b128
  return u.v;
}

// B-matrix 32x16 f16 fragment: lane l holds column N=l%16,
// K = (l<16 ? 0..7,16..23 : 8..15,24..31), matching the A fragment.
// Weights staged in LDS transposed: Wt[N][Kpad], K contiguous.
__device__ __forceinline__ v16h load_B_frag(const _Float16* wt, int kStride,
                                            int lane, int kChunk, int nTile) {
  const int n = nTile * 16 + (lane & 15);
  const _Float16* p = wt + n * kStride + kChunk * 32 + ((lane & 16) ? 8 : 0);
  union { v16h v; v8h h[2]; } u;
  u.h[0] = *(const v8h*)(p);
  u.h[1] = *(const v8h*)(p + 16);
  return u.v;
}

// One dense layer for a 16-node tile: out[16][NTILES*16] = tanh(in @ W + b)
// D-layout (f32 16x16): reg r, lane l -> M = r + (l>=16 ? 8:0), N = l%16
template<int KCHUNKS, int NTILES>
__device__ __forceinline__ void mlp_layer(const _Float16* in, _Float16* out,
                                          const _Float16* wt, int kStride,
                                          const float* bias, int lane) {
  v16h a[KCHUNKS];
#pragma unroll
  for (int k = 0; k < KCHUNKS; ++k) a[k] = load_A_frag(in, lane, k);
  const int n = lane & 15;
  const int mbase = (lane & 16) ? 8 : 0;
#pragma unroll
  for (int t = 0; t < NTILES; ++t) {
    v8f acc = {};
#pragma unroll
    for (int k = 0; k < KCHUNKS; ++k) {
      v16h b = load_B_frag(wt, kStride, lane, k, t);
      acc = __builtin_amdgcn_wmma_f32_16x16x32_f16(false, a[k], false, b,
                                                   (short)0, acc, false, false);
    }
    const float bv = bias[t * 16 + n];
#pragma unroll
    for (int r = 0; r < 8; ++r) {
      float y = fast_tanh(acc[r] + bv);
      out[(mbase + r) * TS + t * 16 + n] = (_Float16)y;
    }
  }
}

// Stage W[fanIn][fanOut] (f32, row-major) into LDS as f16 Wt[Npad][Kpad],
// transposed (Wt[n][k] = W[k][n]); padding = 0 so padded WMMAs are exact.
__device__ __forceinline__ void load_wt(const float* W, _Float16* wt,
                                        int fanIn, int fanOut, int Kpad, int Npad,
                                        int tid, int nthreads) {
  for (int i = tid; i < Npad * Kpad; i += nthreads) {
    const int nn = i / Kpad, kk = i % Kpad;
    const float v = (nn < fanOut && kk < fanIn) ? W[kk * fanOut + nn] : 0.0f;
    wt[i] = (_Float16)v;
  }
}

__global__ __launch_bounds__(WAVES * 32)
void gcnet_mlp_kernel(const float* __restrict__ x,
                      const float* __restrict__ W1, const float* __restrict__ b1,
                      const float* __restrict__ W2, const float* __restrict__ b2,
                      const float* __restrict__ W3, const float* __restrict__ b3,
                      const float* __restrict__ W4, const float* __restrict__ b4,
                      const float* __restrict__ W5, const float* __restrict__ b5,
                      const float* __restrict__ Wl, const float* __restrict__ bl,
                      float* __restrict__ out, int n_nodes) {
  __shared__ __align__(32) _Float16 sW1[16 * 32];   // 3->16,  Kpad 32
  __shared__ __align__(32) _Float16 sW2[32 * 32];   // 16->32, Kpad 32
  __shared__ __align__(32) _Float16 sW3[64 * 32];   // 32->64, Kpad 32
  __shared__ __align__(32) _Float16 sW4[32 * 64];   // 64->32, Kpad 64
  __shared__ __align__(32) _Float16 sW5[16 * 32];   // 32->16, Kpad 32
  __shared__ __align__(32) _Float16 sWl[16 * 32];   // 16->3,  Kpad 32, Npad 16
  __shared__ __align__(16) float    sB[176];        // biases, padded per-tile
  __shared__ __align__(32) _Float16 sBufA[WAVES][16 * TS];
  __shared__ __align__(32) _Float16 sBufB[WAVES][16 * TS];

  const int tid = threadIdx.x;
  const int nthreads = WAVES * 32;

  load_wt(W1, sW1,  3, 16, 32, 16, tid, nthreads);
  load_wt(W2, sW2, 16, 32, 32, 32, tid, nthreads);
  load_wt(W3, sW3, 32, 64, 32, 64, tid, nthreads);
  load_wt(W4, sW4, 64, 32, 64, 32, tid, nthreads);
  load_wt(W5, sW5, 32, 16, 32, 16, tid, nthreads);
  load_wt(Wl, sWl, 16,  3, 32, 16, tid, nthreads);
  for (int i = tid; i < 176; i += nthreads) {
    float v = 0.0f;
    if      (i <  16) v = b1[i];
    else if (i <  48) v = b2[i - 16];
    else if (i < 112) v = b3[i - 48];
    else if (i < 144) v = b4[i - 112];
    else if (i < 160) v = b5[i - 144];
    else if (i - 160 < 3) v = bl[i - 160];
    sB[i] = v;
  }
  __syncthreads();

  const int  lane = tid & 31;
  const int  wave = tid >> 5;
  const long tile = (long)blockIdx.x * WAVES + wave;
  const long nodeBase = tile * 16;
  if (nodeBase >= n_nodes) return;

  _Float16* bufA = sBufA[wave];
  _Float16* bufB = sBufB[wave];

  // zero both wave-private tiles (so K/N padding columns read as 0)
  {
    int4 z = make_int4(0, 0, 0, 0);
    int4* pa = (int4*)bufA;
    int4* pb = (int4*)bufB;
    for (int i = lane; i < (16 * TS) / 8; i += 32) { pa[i] = z; pb[i] = z; }
  }
  WSYNC();

  // load 16 nodes x 3 coords (48 contiguous f32) into bufA cols 0..2
  {
    const long base = nodeBase * 3;
    for (int i = lane; i < 48; i += 32) {
      const float v = (base + i < (long)n_nodes * 3) ? x[base + i] : 0.0f;
      bufA[(i / 3) * TS + (i % 3)] = (_Float16)v;
    }
  }
  WSYNC();

  mlp_layer<1, 1>(bufA, bufB, sW1, 32, sB + 0,   lane); WSYNC();  // -> 16
  mlp_layer<1, 2>(bufB, bufA, sW2, 32, sB + 16,  lane); WSYNC();  // -> 32
  mlp_layer<1, 4>(bufA, bufB, sW3, 32, sB + 48,  lane); WSYNC();  // -> 64
  mlp_layer<2, 2>(bufB, bufA, sW4, 64, sB + 112, lane); WSYNC();  // -> 32
  mlp_layer<1, 1>(bufA, bufB, sW5, 32, sB + 144, lane);           // -> 16

  // bufB cols 16..31 hold stale Y3 data; zero them so the final layer's
  // K-padding (16 -> 32) contributes nothing.
  {
    const int n = lane & 15;
    const int mbase = (lane & 16) ? 8 : 0;
#pragma unroll
    for (int r = 0; r < 8; ++r)
      bufB[(mbase + r) * TS + 16 + n] = (_Float16)0.0f;
  }
  WSYNC();

  // final linear layer (no activation), N padded 3->16, write f32 out
  {
    v16h a0 = load_A_frag(bufB, lane, 0);
    v16h b0 = load_B_frag(sWl, 32, lane, 0, 0);
    v8f acc = {};
    acc = __builtin_amdgcn_wmma_f32_16x16x32_f16(false, a0, false, b0,
                                                 (short)0, acc, false, false);
    const int n = lane & 15;
    const int mbase = (lane & 16) ? 8 : 0;
    if (n < 3) {
      const float bv = sB[160 + n];
#pragma unroll
      for (int r = 0; r < 8; ++r) {
        const long node = nodeBase + mbase + r;
        if (node < n_nodes) out[node * 3 + n] = acc[r] + bv;
      }
    }
  }
}

extern "C" void kernel_launch(void* const* d_in, const int* in_sizes, int n_in,
                              void* d_out, int out_size, void* d_ws, size_t ws_size,
                              hipStream_t stream) {
  const float* x  = (const float*)d_in[0];
  // d_in[1] = edge_index (unused: ChebConv K=1 has no propagation term)
  const float* W1 = (const float*)d_in[2];
  const float* b1 = (const float*)d_in[3];
  const float* W2 = (const float*)d_in[4];
  const float* b2 = (const float*)d_in[5];
  const float* W3 = (const float*)d_in[6];
  const float* b3 = (const float*)d_in[7];
  const float* W4 = (const float*)d_in[8];
  const float* b4 = (const float*)d_in[9];
  const float* W5 = (const float*)d_in[10];
  const float* b5 = (const float*)d_in[11];
  const float* Wl = (const float*)d_in[12];
  const float* bl = (const float*)d_in[13];
  float* out = (float*)d_out;

  const int n_nodes = in_sizes[0] / 3;
  const int tiles   = (n_nodes + 15) / 16;
  const int blocks  = (tiles + WAVES - 1) / WAVES;

  gcnet_mlp_kernel<<<blocks, WAVES * 32, 0, stream>>>(
      x, W1, b1, W2, b2, W3, b3, W4, b4, W5, b5, Wl, bl, out, n_nodes);
}
